// MultiHeadAttention_15410342658150
// MI455X (gfx1250) — compile-verified, matrix operand order checked by reading, not run on hardware
//
#include <hip/hip_runtime.h>

// MI455X / gfx1250 multi-head attention forward.
// - All GEMMs: v_wmma_f32_16x16x32_f16, f32 accumulate, 2x2 tiles per wave
//   (block tile 128x64, 8 waves as 4Mx2N, 4 WMMA per K-step).
// - Pure f16 global->LDS staging uses global_load_async_to_lds_b128
//   (ASYNCcnt) so the async engine feeds LDS without a VGPR round trip.
// - Softmax in f32. Attention probabilities live in d_out (mandated output)
//   and are re-read by the PV stage.

#define DIM   1024
#define NHEAD 16
#define DKH   64
#define BATCH 2
#define SEQ   2048
#define MROWS (BATCH * SEQ)   // 4096

typedef __attribute__((ext_vector_type(4)))  _Float16 v4h;
typedef __attribute__((ext_vector_type(8)))  _Float16 v8h;
typedef __attribute__((ext_vector_type(16))) _Float16 v16h;
typedef __attribute__((ext_vector_type(8)))  float    v8f;

// LDS tile row stride in halves: 32 data + 8 pad = 80 bytes/row.
#define LDS_STRIDE 40

// ---- CDNA5 async global->LDS copy (16 bytes per lane), ASYNCcnt-tracked ----
__device__ __forceinline__ void async_copy_b128(void* lds_ptr, const void* gptr) {
  const uint32_t lds_off = (uint32_t)(uintptr_t)lds_ptr;   // addr[31:0] = LDS byte addr
  const uint64_t gaddr   = (uint64_t)(uintptr_t)gptr;
  asm volatile("global_load_async_to_lds_b128 %0, %1, off"
               :: "v"(lds_off), "v"(gaddr)
               : "memory");
}
__device__ __forceinline__ void wait_async0() {
  asm volatile("s_wait_asynccnt 0x0" ::: "memory");
}

// Load one 16x32 f16 WMMA operand fragment from a row-major LDS tile.
// ISA 7.12.2: lane = m + 16*khalf holds K = kb..kb+7 and kb+16..kb+23
// (kb = 8*khalf) -> two contiguous 16-byte LDS loads per lane. B (32x16) is
// the mirror image with lane = n, so row-major-along-K tiles serve both.
__device__ __forceinline__ v16h frag_load(const _Float16* __restrict__ tile, int lane) {
  const int m  = lane & 15;
  const int kb = (lane >> 4) << 3;
  const _Float16* p = tile + m * LDS_STRIDE + kb;
  v8h lo = *(const v8h*)(p);
  v8h hi = *(const v8h*)(p + 16);
  v16h f;
#pragma unroll
  for (int i = 0; i < 8; ++i) { f[i] = lo[i]; f[i + 8] = hi[i]; }
  return f;
}

#define WMMA_F16(a, b, c) \
  __builtin_amdgcn_wmma_f32_16x16x32_f16(false, (a), false, (b), (short)0, (c), false, false)

// ---------------------------------------------------------------------------
// Y = (X @ W^T + b) * scale over [MROWS, DIM] x [DIM, DIM].
// Block 256 thr / 8 waves -> 128x64 tile; wave (wm = w&3, wn = w>>2) owns a
// 32x32 patch = 2x2 WMMA tiles. K stepped by 32.
// ---------------------------------------------------------------------------
template <bool A_F16, bool OUT_F16>
__global__ __launch_bounds__(256)
void gemm_xWt(const void* __restrict__ Aany, const float* __restrict__ W,
              const float* __restrict__ bias, void* __restrict__ Outany,
              float scale) {
  __shared__ __align__(16) _Float16 As[128 * LDS_STRIDE];
  __shared__ __align__(16) _Float16 Bs[64 * LDS_STRIDE];

  const int tid  = threadIdx.x;
  const int wave = tid >> 5;
  const int lane = tid & 31;
  const int wm   = wave & 3;        // M strip (32 rows)
  const int wn   = wave >> 2;       // N strip (32 cols)
  const int m0   = blockIdx.x * 128;
  const int n0   = blockIdx.y * 64;

  v8f acc00 = {}, acc01 = {}, acc10 = {}, acc11 = {};

  for (int k0 = 0; k0 < DIM; k0 += 32) {
    // ---- stage A tile: 128 x 32 ----
    if constexpr (A_F16) {
      const _Float16* A = (const _Float16*)Aany;
#pragma unroll
      for (int i = 0; i < 2; ++i) {
        int cidx = tid + i * 256;            // 512 chunks of 8 halves
        int row  = cidx >> 2;
        int c8   = (cidx & 3) << 3;
        async_copy_b128(As + row * LDS_STRIDE + c8,
                        A + (size_t)(m0 + row) * DIM + k0 + c8);
      }
    } else {
      const float* A = (const float*)Aany;
#pragma unroll
      for (int i = 0; i < 4; ++i) {
        int idx = tid + i * 256;             // 1024 chunks of 4 floats
        int row = idx >> 3;
        int c4  = (idx & 7) << 2;
        float4 g = *(const float4*)(A + (size_t)(m0 + row) * DIM + k0 + c4);
        v4h h;
        h[0] = (_Float16)g.x; h[1] = (_Float16)g.y;
        h[2] = (_Float16)g.z; h[3] = (_Float16)g.w;
        *(v4h*)(As + row * LDS_STRIDE + c4) = h;
      }
    }
    // ---- stage B tile: W rows n0..n0+63, cols k0..k0+31 (f32 -> f16) ----
    {
#pragma unroll
      for (int i = 0; i < 2; ++i) {
        int idx = tid + i * 256;             // 512 chunks of 4 floats
        int row = idx >> 3;
        int c4  = (idx & 7) << 2;
        float4 g = *(const float4*)(W + (size_t)(n0 + row) * DIM + k0 + c4);
        v4h h;
        h[0] = (_Float16)g.x; h[1] = (_Float16)g.y;
        h[2] = (_Float16)g.z; h[3] = (_Float16)g.w;
        *(v4h*)(Bs + row * LDS_STRIDE + c4) = h;
      }
    }
    if constexpr (A_F16) wait_async0();
    __syncthreads();

    v16h a0 = frag_load(As + (wm * 32 +  0) * LDS_STRIDE, lane);
    v16h a1 = frag_load(As + (wm * 32 + 16) * LDS_STRIDE, lane);
    v16h b0 = frag_load(Bs + (wn * 32 +  0) * LDS_STRIDE, lane);
    v16h b1 = frag_load(Bs + (wn * 32 + 16) * LDS_STRIDE, lane);
    acc00 = WMMA_F16(a0, b0, acc00);
    acc01 = WMMA_F16(a0, b1, acc01);
    acc10 = WMMA_F16(a1, b0, acc10);
    acc11 = WMMA_F16(a1, b1, acc11);
    __syncthreads();
  }

  // ---- epilogue ----
  const int n  = lane & 15;
  const int mb = (lane >> 4) << 3;
  const float bn0 = bias[n0 + wn * 32 + n];
  const float bn1 = bias[n0 + wn * 32 + 16 + n];
#pragma unroll
  for (int r = 0; r < 8; ++r) {
    const int mA = m0 + wm * 32 + mb + r;
    const int mB = mA + 16;
    const int cA = n0 + wn * 32 + n;
    const int cB = cA + 16;
    float v00 = (acc00[r] + bn0) * scale;
    float v01 = (acc01[r] + bn1) * scale;
    float v10 = (acc10[r] + bn0) * scale;
    float v11 = (acc11[r] + bn1) * scale;
    if constexpr (OUT_F16) {
      _Float16* O = (_Float16*)Outany;
      O[(size_t)mA * DIM + cA] = (_Float16)v00;
      O[(size_t)mA * DIM + cB] = (_Float16)v01;
      O[(size_t)mB * DIM + cA] = (_Float16)v10;
      O[(size_t)mB * DIM + cB] = (_Float16)v11;
    } else {
      float* O = (float*)Outany;
      O[(size_t)mA * DIM + cA] = v00;
      O[(size_t)mA * DIM + cB] = v01;
      O[(size_t)mB * DIM + cA] = v10;
      O[(size_t)mB * DIM + cB] = v11;
    }
  }
}

// ---------------------------------------------------------------------------
// scores[b,h,q,kk] = sum_d Qh[b,q,h*64+d] * Kh[b,kk,h*64+d]   (scale in Qh)
// Block tile 128(q) x 64(kk); K = 64 (2 steps). Q/K tiles staged with the
// async global->LDS engine.
// ---------------------------------------------------------------------------
__global__ __launch_bounds__(256)
void scores_kernel(const _Float16* __restrict__ Qh,
                   const _Float16* __restrict__ Kh,
                   float* __restrict__ attn) {
  __shared__ __align__(16) _Float16 As[128 * LDS_STRIDE];
  __shared__ __align__(16) _Float16 Bs[64 * LDS_STRIDE];

  const int tid  = threadIdx.x;
  const int wave = tid >> 5;
  const int lane = tid & 31;
  const int wm   = wave & 3;
  const int wn   = wave >> 2;
  const int z    = blockIdx.z;          // b*NHEAD + h
  const int b    = z / NHEAD;
  const int h    = z % NHEAD;
  const int q0   = blockIdx.x * 128;
  const int kk0  = blockIdx.y * 64;

  v8f acc00 = {}, acc01 = {}, acc10 = {}, acc11 = {};

#pragma unroll
  for (int k0 = 0; k0 < DKH; k0 += 32) {
    // Q tile 128x32: 512 b128 chunks, async
#pragma unroll
    for (int i = 0; i < 2; ++i) {
      int cidx = tid + i * 256;
      int row  = cidx >> 2;
      int c8   = (cidx & 3) << 3;
      async_copy_b128(As + row * LDS_STRIDE + c8,
                      Qh + (size_t)(b * SEQ + q0 + row) * DIM + h * DKH + k0 + c8);
    }
    // K tile 64x32: 256 b128 chunks, async
    {
      int row = tid >> 2;
      int c8  = (tid & 3) << 3;
      async_copy_b128(Bs + row * LDS_STRIDE + c8,
                      Kh + (size_t)(b * SEQ + kk0 + row) * DIM + h * DKH + k0 + c8);
    }
    wait_async0();
    __syncthreads();

    v16h a0 = frag_load(As + (wm * 32 +  0) * LDS_STRIDE, lane);
    v16h a1 = frag_load(As + (wm * 32 + 16) * LDS_STRIDE, lane);
    v16h b0 = frag_load(Bs + (wn * 32 +  0) * LDS_STRIDE, lane);
    v16h b1 = frag_load(Bs + (wn * 32 + 16) * LDS_STRIDE, lane);
    acc00 = WMMA_F16(a0, b0, acc00);
    acc01 = WMMA_F16(a0, b1, acc01);
    acc10 = WMMA_F16(a1, b0, acc10);
    acc11 = WMMA_F16(a1, b1, acc11);
    __syncthreads();
  }

  const int n  = lane & 15;
  const int mb = (lane >> 4) << 3;
#pragma unroll
  for (int r = 0; r < 8; ++r) {
    const int qA = q0 + wm * 32 + mb + r;
    const int qB = qA + 16;
    const int cA = kk0 + wn * 32 + n;
    const int cB = cA + 16;
    attn[((size_t)z * SEQ + qA) * SEQ + cA] = acc00[r];
    attn[((size_t)z * SEQ + qA) * SEQ + cB] = acc01[r];
    attn[((size_t)z * SEQ + qB) * SEQ + cA] = acc10[r];
    attn[((size_t)z * SEQ + qB) * SEQ + cB] = acc11[r];
  }
}

// ---------------------------------------------------------------------------
// Row softmax over attn in place. One 256-thread block per row of 2048.
// ---------------------------------------------------------------------------
__global__ __launch_bounds__(256)
void softmax_kernel(float* __restrict__ attn) {
  __shared__ float red[256];
  const int t = threadIdx.x;
  float* row = attn + (size_t)blockIdx.x * SEQ;

  float4 a = *(const float4*)(row + t * 8);
  float4 b = *(const float4*)(row + t * 8 + 4);

  float m = fmaxf(fmaxf(fmaxf(a.x, a.y), fmaxf(a.z, a.w)),
                  fmaxf(fmaxf(b.x, b.y), fmaxf(b.z, b.w)));
  red[t] = m;
  __syncthreads();
  for (int s = 128; s > 0; s >>= 1) {
    if (t < s) red[t] = fmaxf(red[t], red[t + s]);
    __syncthreads();
  }
  m = red[0];
  __syncthreads();

  a.x = __expf(a.x - m); a.y = __expf(a.y - m);
  a.z = __expf(a.z - m); a.w = __expf(a.w - m);
  b.x = __expf(b.x - m); b.y = __expf(b.y - m);
  b.z = __expf(b.z - m); b.w = __expf(b.w - m);

  red[t] = (a.x + a.y + a.z + a.w) + (b.x + b.y + b.z + b.w);
  __syncthreads();
  for (int s = 128; s > 0; s >>= 1) {
    if (t < s) red[t] += red[t + s];
    __syncthreads();
  }
  const float inv = 1.0f / red[0];

  a.x *= inv; a.y *= inv; a.z *= inv; a.w *= inv;
  b.x *= inv; b.y *= inv; b.z *= inv; b.w *= inv;
  *(float4*)(row + t * 8)     = a;
  *(float4*)(row + t * 8 + 4) = b;
}

// ---------------------------------------------------------------------------
// ctx[b,q,h*64+d] = sum_kk P[b,h,q,kk] * Vh[b,kk,h*64+d]
// Block tile 128(q) x 64(d) covers the whole head dim; K = SEQ (64 steps).
// P converted f32->f16 on staging; V staged transposed.
// ---------------------------------------------------------------------------
__global__ __launch_bounds__(256)
void pv_kernel(const float* __restrict__ attn, const _Float16* __restrict__ Vh,
               _Float16* __restrict__ ctx) {
  __shared__ __align__(16) _Float16 As[128 * LDS_STRIDE];
  __shared__ __align__(16) _Float16 Bs[64 * LDS_STRIDE];   // Bs[n][kk]

  const int tid  = threadIdx.x;
  const int wave = tid >> 5;
  const int lane = tid & 31;
  const int wm   = wave & 3;
  const int wn   = wave >> 2;
  const int z    = blockIdx.z;
  const int b    = z / NHEAD;
  const int h    = z % NHEAD;
  const int q0   = blockIdx.x * 128;

  v8f acc00 = {}, acc01 = {}, acc10 = {}, acc11 = {};

  for (int k0 = 0; k0 < SEQ; k0 += 32) {
    // P tile 128x32, f32 -> f16
#pragma unroll
    for (int i = 0; i < 4; ++i) {
      int idx = tid + i * 256;
      int row = idx >> 3;
      int c4  = (idx & 7) << 2;
      float4 g = *(const float4*)(attn + ((size_t)z * SEQ + q0 + row) * SEQ + k0 + c4);
      v4h hv;
      hv[0] = (_Float16)g.x; hv[1] = (_Float16)g.y;
      hv[2] = (_Float16)g.z; hv[3] = (_Float16)g.w;
      *(v4h*)(As + row * LDS_STRIDE + c4) = hv;
    }
    // V tile transposed: Bs[n][kk] = V[k0+kk][h*64+n], n in 0..63
#pragma unroll
    for (int i = 0; i < 2; ++i) {
      int idx = tid + i * 256;            // 512 v4h chunks
      int kk  = idx >> 4;                 // 0..31
      int nb  = (idx & 15) << 2;          // 0..60
      v4h v = *(const v4h*)(Vh + (size_t)(b * SEQ + k0 + kk) * DIM + h * DKH + nb);
#pragma unroll
      for (int j = 0; j < 4; ++j)
        Bs[(nb + j) * LDS_STRIDE + kk] = v[j];
    }
    __syncthreads();

    v16h a0 = frag_load(As + (wm * 32 +  0) * LDS_STRIDE, lane);
    v16h a1 = frag_load(As + (wm * 32 + 16) * LDS_STRIDE, lane);
    v16h b0 = frag_load(Bs + (wn * 32 +  0) * LDS_STRIDE, lane);
    v16h b1 = frag_load(Bs + (wn * 32 + 16) * LDS_STRIDE, lane);
    acc00 = WMMA_F16(a0, b0, acc00);
    acc01 = WMMA_F16(a0, b1, acc01);
    acc10 = WMMA_F16(a1, b0, acc10);
    acc11 = WMMA_F16(a1, b1, acc11);
    __syncthreads();
  }

  const int n  = lane & 15;
  const int mb = (lane >> 4) << 3;
#pragma unroll
  for (int r = 0; r < 8; ++r) {
    const int qA = q0 + wm * 32 + mb + r;
    const int qB = qA + 16;
    const int cA = h * DKH + wn * 32 + n;
    const int cB = cA + 16;
    ctx[(size_t)(b * SEQ + qA) * DIM + cA] = (_Float16)acc00[r];
    ctx[(size_t)(b * SEQ + qA) * DIM + cB] = (_Float16)acc01[r];
    ctx[(size_t)(b * SEQ + qB) * DIM + cA] = (_Float16)acc10[r];
    ctx[(size_t)(b * SEQ + qB) * DIM + cB] = (_Float16)acc11[r];
  }
}

// ---------------------------------------------------------------------------
extern "C" void kernel_launch(void* const* d_in, const int* in_sizes, int n_in,
                              void* d_out, int out_size, void* d_ws, size_t ws_size,
                              hipStream_t stream) {
  const float* query = (const float*)d_in[0];
  const float* key_  = (const float*)d_in[1];
  const float* value = (const float*)d_in[2];
  const float* wq    = (const float*)d_in[3];
  const float* bq    = (const float*)d_in[4];
  const float* wk    = (const float*)d_in[5];
  const float* bk    = (const float*)d_in[6];
  const float* wv    = (const float*)d_in[7];
  const float* bv    = (const float*)d_in[8];
  const float* wo    = (const float*)d_in[9];
  const float* bo    = (const float*)d_in[10];

  float* out  = (float*)d_out;                          // [B,S,D]
  float* attn = out + (size_t)MROWS * DIM;              // [B,H,S,S]

  const size_t qkv_elems = (size_t)MROWS * DIM;
  _Float16* Qh  = (_Float16*)d_ws;
  _Float16* Kh  = Qh + qkv_elems;
  _Float16* Vh  = Kh + qkv_elems;
  _Float16* Ctx = Vh + qkv_elems;                       // 32 MB of ws total

  const dim3 blk(256);
  const dim3 gproj(MROWS / 128, DIM / 64);              // 32 x 16

  // Q/K/V projections (Q pre-scaled by 1/sqrt(dk) = 0.125)
  hipLaunchKernelGGL((gemm_xWt<false, true>), gproj, blk, 0, stream,
                     (const void*)query, wq, bq, (void*)Qh, 0.125f);
  hipLaunchKernelGGL((gemm_xWt<false, true>), gproj, blk, 0, stream,
                     (const void*)key_, wk, bk, (void*)Kh, 1.0f);
  hipLaunchKernelGGL((gemm_xWt<false, true>), gproj, blk, 0, stream,
                     (const void*)value, wv, bv, (void*)Vh, 1.0f);

  // scores = Q K^T
  const dim3 gsc(SEQ / 128, SEQ / 64, BATCH * NHEAD);   // 16 x 32 x 32
  hipLaunchKernelGGL(scores_kernel, gsc, blk, 0, stream, Qh, Kh, attn);

  // row softmax in place (== "attention" output)
  hipLaunchKernelGGL(softmax_kernel, dim3(BATCH * NHEAD * SEQ), blk, 0, stream, attn);

  // context = P V
  const dim3 gpv(SEQ / 128, 1, BATCH * NHEAD);          // 16 x 1 x 32
  hipLaunchKernelGGL(pv_kernel, gpv, blk, 0, stream, attn, Vh, Ctx);

  // out = ctx W_o^T + b_o
  hipLaunchKernelGGL((gemm_xWt<true, false>), gproj, blk, 0, stream,
                     (const void*)Ctx, wo, bo, (void*)out, 1.0f);
}
